// Model_17755394801725
// MI455X (gfx1250) — compile-verified, matrix operand order checked by reading, not run on hardware
//
#include <hip/hip_runtime.h>
#include <hip/hip_bf16.h>

typedef unsigned short u16;
typedef __attribute__((ext_vector_type(16))) __bf16 v16bf;
typedef __attribute__((ext_vector_type(8)))  float  v8f;

#define T_LEN 256
#define BATCH 64
#define D_EMB 256
#define D_MLP 512
#define D_RNN 1024
#define D_LAT 256
#define ROWS  (T_LEN * BATCH)     // 16384
#define LOG2PI_F 1.8378770664093453f

// ---------- scalar helpers ----------
__device__ __forceinline__ u16 f2bf(float f) {
  unsigned int u = __float_as_uint(f);
  unsigned int r = (u + 0x7FFFu + ((u >> 16) & 1u)) >> 16;
  return (u16)r;
}
__device__ __forceinline__ float sigm(float x) { return 1.f / (1.f + expf(-x)); }
__device__ __forceinline__ float lrelu(float x) {
  float v = x > 0.f ? x : x * (1.f / 3.f);
  return fminf(3.f, fmaxf(-3.f, v));
}
__device__ __forceinline__ float clip8(float x) { return fminf(8.f, fmaxf(-8.f, x)); }

// ---------- WMMA fragment helpers (ISA 7.12.2 layouts) ----------
union FragAB { v16bf v; uint4 u[2]; };

// bf16 16x32 frag: lane<16 holds K {k0..k0+7, k0+16..k0+23},
// lane>=16 holds K {k0+8..k0+15, k0+24..k0+31}; row = lane&15.
__device__ __forceinline__ v16bf ld_frag(const u16* rowbase, int ld, int k0, int lane) {
  int r = lane & 15;
  int h = (lane >> 4) << 3;               // 0 or 8
  const u16* p = rowbase + (size_t)r * ld + k0 + h;
  FragAB f;
  f.u[0] = *(const uint4*)(p);
  f.u[1] = *(const uint4*)(p + 16);
  return f.v;
}

// M-strip (64 rows = 4 tiles) accumulate: c[s] += A(m0+16s:,:) x B(n0:,:)^T,
// one B fragment per k-step shared across the 4 WMMAs (4x weight-reuse).
__device__ __forceinline__ void mm_strip(v8f* c, const u16* A, int lda,
                                         const u16* B, int ldb,
                                         int m0, int n0, int K, int kOffB, int lane) {
  const u16* Ar = A + (size_t)m0 * lda;
  const u16* Br = B + (size_t)n0 * ldb + kOffB;
  for (int k0 = 0; k0 < K; k0 += 32) {
    __builtin_prefetch((const void*)(Br + (size_t)(lane & 15) * ldb + k0 + 128), 0, 1);
    v16bf b = ld_frag(Br, ldb, k0, lane);
#pragma unroll
    for (int s = 0; s < 4; ++s) {
      v16bf a = ld_frag(Ar + (size_t)(s << 4) * lda, lda, k0, lane);
      c[s] = __builtin_amdgcn_wmma_f32_16x16x32_bf16(false, a, false, b, (short)0,
                                                     c[s], false, false);
    }
  }
}

// ---------- small utility kernels ----------
__global__ void k_f32_to_bf16(const float* __restrict__ s, u16* __restrict__ d, int n) {
  for (int i = blockIdx.x * blockDim.x + threadIdx.x; i < n; i += gridDim.x * blockDim.x)
    d[i] = f2bf(s[i]);
}
__global__ void k_add_vec(const float* __restrict__ a, const float* __restrict__ b,
                          float* __restrict__ o, int n) {
  int i = blockIdx.x * blockDim.x + threadIdx.x;
  if (i < n) o[i] = a[i] + b[i];
}
__global__ void k_build_wout(const float* __restrict__ Wcls, const float* __restrict__ Wm1,
                             const float* __restrict__ Wm2, u16* __restrict__ Wout) {
  int i = blockIdx.x * blockDim.x + threadIdx.x;   // over 128*1024
  if (i >= 128 * 1024) return;
  int row = i >> 10, k = i & 1023;
  float v = 0.f;
  if (row == 0)       v = Wcls[k];
  else if (row < 61)  v = Wm1[(row - 1) * 1024 + k];
  else if (row < 121) v = Wm2[(row - 61) * 1024 + k];
  Wout[i] = f2bf(v);
}
__global__ void k_build_bout(const float* __restrict__ bcls, const float* __restrict__ bm1,
                             const float* __restrict__ bm2, float* __restrict__ bout) {
  int i = blockIdx.x * blockDim.x + threadIdx.x;
  if (i >= 128) return;
  float v = 0.f;
  if (i == 0)       v = bcls[0];
  else if (i < 61)  v = bm1[i - 1];
  else if (i < 121) v = bm2[i - 61];
  bout[i] = v;
}

// embeddings: x_emb -> xz[:, :256] (bf16), y_emb time-flipped -> yef (bf16)
__global__ void k_embed(const float* __restrict__ x, const float* __restrict__ y,
                        const float* __restrict__ W_emb, const float* __restrict__ b_emb,
                        u16* __restrict__ yef, u16* __restrict__ xz) {
  int idx = blockIdx.x * blockDim.x + threadIdx.x;   // T*B*256
  if (idx >= ROWS * D_EMB) return;
  int n = idx & 255, row = idx >> 8;
  int t = row / BATCH, b = row % BATCH;
  const float* w = W_emb + n * 3;
  float x0 = x[row * 3], x1 = x[row * 3 + 1], x2 = x[row * 3 + 2];
  float y0 = y[row * 3], y1 = y[row * 3 + 1], y2 = y[row * 3 + 2];
  float xe = x0 * w[0] + x1 * w[1] + x2 * w[2] + b_emb[n];
  float ye = y0 * w[0] + y1 * w[1] + y2 * w[2] + b_emb[n];
  xz[(size_t)row * 768 + n] = f2bf(xe);
  yef[((size_t)(T_LEN - 1 - t) * BATCH + b) * 256 + n] = f2bf(ye);
}

// ---------- dual-input WMMA GEMM (M-strip of 64): C = A1*B1^T + A2*B2^T + bias ----------
__global__ void __launch_bounds__(256) k_wmma_gemm(
    float* __restrict__ C, int ldc, int M, int N,
    const u16* __restrict__ A1, int lda1, const u16* __restrict__ B1, int ldb1, int K1,
    const u16* __restrict__ A2, int lda2, const u16* __restrict__ B2, int ldb2, int K2,
    const float* __restrict__ bias, const float* __restrict__ Cin, int ldcin) {
  int wave = (blockIdx.x * blockDim.x + threadIdx.x) >> 5;
  int lane = threadIdx.x & 31;
  int tn = N >> 4;
  int strips = M >> 6;                        // M multiple of 64
  if (wave >= strips * tn) return;            // wave-uniform
  int m0 = (wave / tn) << 6, n0 = (wave % tn) << 4;
  int col = lane & 15, rbase = (lane >> 4) << 3;
  v8f c[4];
  if (Cin) {
#pragma unroll
    for (int s = 0; s < 4; ++s)
      for (int r = 0; r < 8; ++r)
        c[s][r] = Cin[(size_t)(m0 + (s << 4) + rbase + r) * ldcin + n0 + col];
  } else {
#pragma unroll
    for (int s = 0; s < 4; ++s)
      for (int r = 0; r < 8; ++r) c[s][r] = 0.f;
  }
  mm_strip(c, A1, lda1, B1, ldb1, m0, n0, K1, 0, lane);
  if (A2) mm_strip(c, A2, lda2, B2, ldb2, m0, n0, K2, 0, lane);
  float bv = bias ? bias[n0 + col] : 0.f;
#pragma unroll
  for (int s = 0; s < 4; ++s)
    for (int r = 0; r < 8; ++r)
      C[(size_t)(m0 + (s << 4) + rbase + r) * ldc + n0 + col] = c[s][r] + bv;
}

// ---------- LSTM pointwise ----------
__global__ void k_lstm_pw(const float* __restrict__ gates, float* __restrict__ cst,
                          u16* __restrict__ hb, u16* __restrict__ hist_bf,
                          float* __restrict__ hist_f32, int row0) {
  int idx = blockIdx.x * blockDim.x + threadIdx.x;
  if (idx >= BATCH * D_RNN) return;
  int b = idx >> 10, n = idx & 1023;
  const float* g = gates + (size_t)b * 4096;
  float gi = g[n], gf = g[1024 + n], gg = g[2048 + n], go = g[3072 + n];
  float c2 = sigm(gf) * cst[idx] + sigm(gi) * tanhf(gg);
  float h2 = sigm(go) * tanhf(c2);
  cst[idx] = c2;
  hb[idx] = f2bf(h2);
  size_t o = (size_t)(row0 + b) * D_RNN + n;
  if (hist_bf)  hist_bf[o]  = f2bf(h2);
  if (hist_f32) hist_f32[o] = h2;
}

// ---------- fused per-step prior/posterior MLP ----------
// One WGP, 32 waves. h and brnn[t] (128KB each) are async-copied into 256KB of
// dynamic LDS (ASYNCcnt path), the 64KB static LDS stages activations.
__global__ void __launch_bounds__(1024) k_fstep_mlp(
    const u16* __restrict__ hb, const u16* __restrict__ br, const float* __restrict__ eps,
    const u16* __restrict__ Wp1b, const float* __restrict__ bp1v,
    const u16* __restrict__ Wp2b, const float* __restrict__ bp2v,
    const u16* __restrict__ Wq1b, const float* __restrict__ bq1v,
    const u16* __restrict__ Wq2b, const float* __restrict__ bq2v,
    const u16* __restrict__ Wgb,  const float* __restrict__ bgv,
    float* __restrict__ pmu, float* __restrict__ plv,
    float* __restrict__ qmu, float* __restrict__ qlv,
    u16* __restrict__ xzt) {
  __shared__ u16 sh[BATCH * D_MLP];           // 64KB activation staging
  extern __shared__ u16 dynsh[];              // 256KB: [0:64K) = h, [64K:128K) = br
  int tid = threadIdx.x, w = tid >> 5, lane = tid & 31;
  int col = lane & 15, rbase = (lane >> 4) << 3;
  const u16* shb = dynsh;
  const u16* sbr = dynsh + 65536;

  // ---- async global->LDS staging of h and br (16B per lane-op) ----
  for (int i = tid; i < 16384; i += 1024) {
    const u16* gsrc = (i < 8192) ? (hb + (size_t)i * 8) : (br + (size_t)(i - 8192) * 8);
    unsigned dst = (unsigned)(size_t)(dynsh + (size_t)i * 8);   // LDS byte offset
    unsigned long long ga = (unsigned long long)(size_t)gsrc;
    asm volatile("global_load_async_to_lds_b128 %0, %1, off"
                 :: "v"(dst), "v"(ga) : "memory");
  }
  asm volatile("s_wait_asynccnt 0x0" ::: "memory");
  __syncthreads();

  int n0 = w << 4;   // 32 waves <-> 32 N-tiles of 512

  // P1: a1 = lrelu(h @ Wp1^T + bp1) -> sh
  {
    v8f c[4];
#pragma unroll
    for (int s = 0; s < 4; ++s) for (int r = 0; r < 8; ++r) c[s][r] = 0.f;
    mm_strip(c, shb, 1024, Wp1b, 1024, 0, n0, 1024, 0, lane);
    float bv = bp1v[n0 + col];
#pragma unroll
    for (int s = 0; s < 4; ++s)
      for (int r = 0; r < 8; ++r)
        sh[((s << 4) + rbase + r) * 512 + n0 + col] = f2bf(lrelu(c[s][r] + bv));
  }
  __syncthreads();
  // P2: pr = clip(a1 @ Wp2^T + bp2) -> pmu/plv
  {
    v8f c[4];
#pragma unroll
    for (int s = 0; s < 4; ++s) for (int r = 0; r < 8; ++r) c[s][r] = 0.f;
    mm_strip(c, (const u16*)sh, 512, Wp2b, 512, 0, n0, 512, 0, lane);
    float bv = bp2v[n0 + col];
    int n = n0 + col;
#pragma unroll
    for (int s = 0; s < 4; ++s)
      for (int r = 0; r < 8; ++r) {
        float v = clip8(c[s][r] + bv);
        int b = (s << 4) + rbase + r;
        if (n < 256) pmu[b * 256 + n] = v; else plv[b * 256 + n - 256] = v;
      }
  }
  __syncthreads();
  // Q1: q1 = lrelu([br,h] @ Wq1^T + bq1) -> sh (K split over concat)
  {
    v8f c[4];
#pragma unroll
    for (int s = 0; s < 4; ++s) for (int r = 0; r < 8; ++r) c[s][r] = 0.f;
    mm_strip(c, sbr, 1024, Wq1b, 2048, 0, n0, 1024, 0,    lane);
    mm_strip(c, shb, 1024, Wq1b, 2048, 0, n0, 1024, 1024, lane);
    float bv = bq1v[n0 + col];
#pragma unroll
    for (int s = 0; s < 4; ++s)
      for (int r = 0; r < 8; ++r)
        sh[((s << 4) + rbase + r) * 512 + n0 + col] = f2bf(lrelu(c[s][r] + bv));
  }
  __syncthreads();
  // Q2: qp = clip(q1 @ Wq2^T + bq2) -> qmu/qlv (global)
  {
    v8f c[4];
#pragma unroll
    for (int s = 0; s < 4; ++s) for (int r = 0; r < 8; ++r) c[s][r] = 0.f;
    mm_strip(c, (const u16*)sh, 512, Wq2b, 512, 0, n0, 512, 0, lane);
    float bv = bq2v[n0 + col];
    int n = n0 + col;
#pragma unroll
    for (int s = 0; s < 4; ++s)
      for (int r = 0; r < 8; ++r) {
        float v = clip8(c[s][r] + bv);
        int b = (s << 4) + rbase + r;
        if (n < 256) qmu[b * 256 + n] = v; else qlv[b * 256 + n - 256] = v;
      }
  }
  __threadfence();
  __syncthreads();
  // Z: z = qmu + eps*exp(0.5*qlv) -> sh cols 0..255
  for (int i = tid; i < BATCH * D_LAT; i += 1024) {
    int b = i >> 8, l = i & 255;
    float z = qmu[b * 256 + l] + eps[b * 256 + l] * expf(0.5f * qlv[b * 256 + l]);
    sh[b * 512 + l] = f2bf(z);
  }
  __syncthreads();
  // G: pz = z @ Wgen^T + bgen -> xz[:, 256:768]
  {
    v8f c[4];
#pragma unroll
    for (int s = 0; s < 4; ++s) for (int r = 0; r < 8; ++r) c[s][r] = 0.f;
    mm_strip(c, (const u16*)sh, 512, Wgb, 256, 0, n0, 256, 0, lane);
    float bv = bgv[n0 + col];
#pragma unroll
    for (int s = 0; s < 4; ++s)
      for (int r = 0; r < 8; ++r)
        xzt[(size_t)((s << 4) + rbase + r) * 768 + 256 + n0 + col] = f2bf(c[s][r] + bv);
  }
}

// ---------- final losses ----------
__device__ float gmm_nll_dev(const float* __restrict__ p, float yv) {
  float mx = -1e30f;
  for (int m = 0; m < 20; ++m) mx = fmaxf(mx, p[m]);
  float s = 0.f;
  for (int m = 0; m < 20; ++m) s += expf(p[m] - mx);
  float lse = mx + logf(s);
  float mx2 = -1e30f;
  for (int m = 0; m < 20; ++m) {
    float d = yv - p[20 + m], lv = p[40 + m];
    float lpm = (p[m] - lse) - 0.5f * (lv + LOG2PI_F + d * d * expf(-lv));
    mx2 = fmaxf(mx2, lpm);
  }
  float s2 = 0.f;
  for (int m = 0; m < 20; ++m) {
    float d = yv - p[20 + m], lv = p[40 + m];
    float lpm = (p[m] - lse) - 0.5f * (lv + LOG2PI_F + d * d * expf(-lv));
    s2 += expf(lpm - mx2);
  }
  return -(mx2 + logf(s2));
}

__global__ void k_final(const float* __restrict__ proj, const float* __restrict__ y,
                        const float* __restrict__ pmus, const float* __restrict__ plvs,
                        const float* __restrict__ qmus, const float* __restrict__ qlvs,
                        float* __restrict__ out_nll, float* __restrict__ out_kld) {
  int r = blockIdx.x * blockDim.x + threadIdx.x;
  if (r >= ROWS) return;
  const float* p = proj + (size_t)r * 128;
  float y0 = y[r * 3], y1 = y[r * 3 + 1], y2 = y[r * 3 + 2];
  float p0 = p[0];
  float l0 = fmaxf(p0, 0.f) + log1pf(expf(-fabsf(p0))) - p0 * y0;
  float l1 = gmm_nll_dev(p + 1,  y1);
  float l2 = gmm_nll_dev(p + 61, y2);
  out_nll[r] = l0 + l1 + l2;
  float ks = 0.f;
  size_t base = (size_t)r * 256;
  for (int l = 0; l < 256; ++l) {
    float pm = pmus[base + l], pl = plvs[base + l];
    float qm = qmus[base + l], ql = qlvs[base + l];
    float d = qm - pm;
    ks += 0.5f * (pl - ql + (expf(ql) + d * d) * expf(-pl) - 1.f);
  }
  out_kld[r] = -ks;
}

// ---------- host orchestration ----------
extern "C" void kernel_launch(void* const* d_in, const int* in_sizes, int n_in,
                              void* d_out, int out_size, void* d_ws, size_t ws_size,
                              hipStream_t stream) {
  (void)in_sizes; (void)n_in; (void)out_size; (void)ws_size;
  const float* x      = (const float*)d_in[0];
  const float* y      = (const float*)d_in[1];
  const float* noises = (const float*)d_in[2];
  const float* W_emb  = (const float*)d_in[3];
  const float* b_emb  = (const float*)d_in[4];
  const float* W_cls  = (const float*)d_in[5];
  const float* b_cls  = (const float*)d_in[6];
  const float* Wih_b  = (const float*)d_in[7];
  const float* Whh_b  = (const float*)d_in[8];
  const float* bih_b  = (const float*)d_in[9];
  const float* bhh_b  = (const float*)d_in[10];
  const float* Wih_f  = (const float*)d_in[11];
  const float* Whh_f  = (const float*)d_in[12];
  const float* bih_f  = (const float*)d_in[13];
  const float* bhh_f  = (const float*)d_in[14];
  const float* W_gen  = (const float*)d_in[15];
  const float* b_gen  = (const float*)d_in[16];
  const float* Wp1    = (const float*)d_in[17];
  const float* bp1    = (const float*)d_in[18];
  const float* Wp2    = (const float*)d_in[19];
  const float* bp2    = (const float*)d_in[20];
  const float* Wq1    = (const float*)d_in[21];
  const float* bq1    = (const float*)d_in[22];
  const float* Wq2    = (const float*)d_in[23];
  const float* bq2    = (const float*)d_in[24];
  const float* Wm1    = (const float*)d_in[25];
  const float* bm1    = (const float*)d_in[26];
  const float* Wm2    = (const float*)d_in[27];
  const float* bm2    = (const float*)d_in[28];

  float* out     = (float*)d_out;
  float* out_nll = out;
  float* out_kld = out + ROWS;
  float* out_hid = out + 2 * ROWS;

  // workspace bump allocator (256B aligned)
  char* wsb = (char*)d_ws;
  size_t off = 0;
  auto alloc = [&](size_t bytes) -> void* {
    off = (off + 255) & ~(size_t)255;
    void* p = wsb + off;
    off += bytes;
    return p;
  };
  u16* Wihb_bf = (u16*)alloc((size_t)4096 * 256  * 2);
  u16* Whhb_bf = (u16*)alloc((size_t)4096 * 1024 * 2);
  u16* Wihf_bf = (u16*)alloc((size_t)4096 * 768  * 2);
  u16* Whhf_bf = (u16*)alloc((size_t)4096 * 1024 * 2);
  u16* Wp1b    = (u16*)alloc((size_t)512 * 1024 * 2);
  u16* Wp2b    = (u16*)alloc((size_t)512 * 512  * 2);
  u16* Wq1b    = (u16*)alloc((size_t)512 * 2048 * 2);
  u16* Wq2b    = (u16*)alloc((size_t)512 * 512  * 2);
  u16* Wgb     = (u16*)alloc((size_t)512 * 256  * 2);
  u16* Woutb   = (u16*)alloc((size_t)128 * 1024 * 2);
  float* bsum_b = (float*)alloc(4096 * 4);
  float* bsum_f = (float*)alloc(4096 * 4);
  float* bout   = (float*)alloc(128 * 4);
  u16* yef   = (u16*)alloc((size_t)ROWS * 256  * 2);
  u16* xz    = (u16*)alloc((size_t)ROWS * 768  * 2);
  u16* brnn  = (u16*)alloc((size_t)ROWS * 1024 * 2);
  u16* hidb  = (u16*)alloc((size_t)ROWS * 1024 * 2);
  u16* hb    = (u16*)alloc((size_t)BATCH * 1024 * 2);
  float* cst   = (float*)alloc((size_t)BATCH * 1024 * 4);
  float* gates = (float*)alloc((size_t)BATCH * 4096 * 4);
  float* pmus = (float*)alloc((size_t)ROWS * 256 * 4);
  float* plvs = (float*)alloc((size_t)ROWS * 256 * 4);
  float* qmus = (float*)alloc((size_t)ROWS * 256 * 4);
  float* qlvs = (float*)alloc((size_t)ROWS * 256 * 4);
  float* proj = (float*)alloc((size_t)ROWS * 128 * 4);

  auto cvt = [&](const float* s, u16* d, int n) {
    int blocks = (n + 255) / 256; if (blocks > 4096) blocks = 4096;
    k_f32_to_bf16<<<blocks, 256, 0, stream>>>(s, d, n);
  };
  cvt(Wih_b, Wihb_bf, 4096 * 256);
  cvt(Whh_b, Whhb_bf, 4096 * 1024);
  cvt(Wih_f, Wihf_bf, 4096 * 768);
  cvt(Whh_f, Whhf_bf, 4096 * 1024);
  cvt(Wp1, Wp1b, 512 * 1024);
  cvt(Wp2, Wp2b, 512 * 512);
  cvt(Wq1, Wq1b, 512 * 2048);
  cvt(Wq2, Wq2b, 512 * 512);
  cvt(W_gen, Wgb, 512 * 256);
  k_build_wout<<<(128 * 1024 + 255) / 256, 256, 0, stream>>>(W_cls, Wm1, Wm2, Woutb);
  k_build_bout<<<1, 128, 0, stream>>>(b_cls, bm1, bm2, bout);
  k_add_vec<<<16, 256, 0, stream>>>(bih_b, bhh_b, bsum_b, 4096);
  k_add_vec<<<16, 256, 0, stream>>>(bih_f, bhh_f, bsum_f, 4096);

  k_embed<<<(ROWS * 256 + 255) / 256, 256, 0, stream>>>(x, y, W_emb, b_emb, yef, xz);

  auto gemm = [&](float* C, int ldc, int M, int N,
                  const u16* A1, int lda1, const u16* B1, int ldb1, int K1,
                  const u16* A2, int lda2, const u16* B2, int ldb2, int K2,
                  const float* bias, const float* Cin, int ldcin) {
    int waves = (M / 64) * (N / 16);
    int blocks = (waves + 7) / 8;
    k_wmma_gemm<<<blocks, 256, 0, stream>>>(C, ldc, M, N, A1, lda1, B1, ldb1, K1,
                                            A2, lda2, B2, ldb2, K2, bias, Cin, ldcin);
  };

  // ---- backward LSTM over flipped sequence ----
  hipMemsetAsync(hb, 0, (size_t)BATCH * 1024 * 2, stream);
  hipMemsetAsync(cst, 0, (size_t)BATCH * 1024 * 4, stream);
  for (int t = 0; t < T_LEN; ++t) {
    gemm(gates, 4096, BATCH, 4096,
         yef + (size_t)t * BATCH * 256, 256, Wihb_bf, 256, 256,
         hb, 1024, Whhb_bf, 1024, 1024,
         bsum_b, nullptr, 0);
    k_lstm_pw<<<(BATCH * 1024 + 255) / 256, 256, 0, stream>>>(
        gates, cst, hb, brnn, nullptr, (T_LEN - 1 - t) * BATCH);
  }

  // ---- forward scan ----
  hipMemsetAsync(hb, 0, (size_t)BATCH * 1024 * 2, stream);
  hipMemsetAsync(cst, 0, (size_t)BATCH * 1024 * 4, stream);
  for (int t = 0; t < T_LEN; ++t) {
    size_t rb = (size_t)t * BATCH;
    k_fstep_mlp<<<1, 1024, 262144, stream>>>(
        hb, brnn + rb * 1024, noises + rb * 256,
        Wp1b, bp1, Wp2b, bp2, Wq1b, bq1, Wq2b, bq2, Wgb, b_gen,
        pmus + rb * 256, plvs + rb * 256, qmus + rb * 256, qlvs + rb * 256,
        xz + rb * 768);
    gemm(gates, 4096, BATCH, 4096,
         xz + rb * 768, 768, Wihf_bf, 768, 768,
         hb, 1024, Whhf_bf, 1024, 1024,
         bsum_f, nullptr, 0);
    k_lstm_pw<<<(BATCH * 1024 + 255) / 256, 256, 0, stream>>>(
        gates, cst, hb, hidb, out_hid, t * BATCH);
  }

  // ---- batched output projection: proj = hidden @ [W_cls;Wm1;Wm2]^T + bias ----
  gemm(proj, 128, ROWS, 128,
       hidb, 1024, Woutb, 1024, 1024,
       nullptr, 0, nullptr, 0, 0,
       bout, nullptr, 0);

  k_final<<<(ROWS + 255) / 256, 256, 0, stream>>>(proj, y, pmus, plvs, qmus, qlvs,
                                                  out_nll, out_kld);
}
